// CustomMoE_2121713845123
// MI455X (gfx1250) — matrix operand order repaired and checked by reading, not compiled, not run on hardware
//
#include <hip/hip_runtime.h>
#include <hip/hip_bf16.h>

// ---------------- problem constants (match reference setup_inputs) ----------
#define T_TOK   8192        // B*S = 4*2048
#define M_DIM   1024
#define H_DIM   4096
#define E_NUM   4
#define K_TOP   2
#define C_CAP   5120        // ceil(1.25 * 2 * 8192 / 4)

typedef __bf16 bf16_t;
typedef bf16_t v16bf __attribute__((ext_vector_type(16)));
typedef float  v8f   __attribute__((ext_vector_type(8)));

union FragU { v16bf v; uint4 q[2]; };

// ---- WMMA fragment loads (per CDNA5 ISA 16-bit layouts) ---------
// A 16x32 (MxK): lanes 0-15 row M=lane, K {0..7,16..23}; lanes 16-31 same row,
// K {8..15, 24..31}.  A is row-major [rows, ldA] bf16.
__device__ inline v16bf load_fragA(const bf16_t* A, int ldA, int row0, int k0, int lane) {
    FragU f;
    const bf16_t* p = A + (size_t)(row0 + (lane & 15)) * ldA + k0 + ((lane >> 4) << 3);
    f.q[0] = *(const uint4*)(p);        // K+0..7   (or K+8..15)
    f.q[1] = *(const uint4*)(p + 16);   // K+16..23 (or K+24..31)
    return f.v;
}
// B 32x16 (KxN): lanes 0-15 hold col N=lane, K {0..7,16..23}; lanes 16-31
// hold K {8..15,24..31}.  W stored N-major: row n = column n of B, length K.
__device__ inline v16bf load_fragB(const bf16_t* W, int ldW, int col0, int k0, int lane) {
    FragU f;
    const bf16_t* p = W + (size_t)(col0 + (lane & 15)) * ldW + k0 + ((lane >> 4) << 3);
    f.q[0] = *(const uint4*)(p);        // K+0..7   (or K+8..15)
    f.q[1] = *(const uint4*)(p + 16);   // K+16..23 (or K+24..31)
    return f.v;
}

// ---------------- 1. gating: logits, softmax, top-2 -------------------------
__global__ __launch_bounds__(256) void gate_kernel(const float* __restrict__ xf,
                                                   const float* __restrict__ wg,
                                                   int* __restrict__ fe, float* __restrict__ gates,
                                                   float* __restrict__ me) {
    const int lane = threadIdx.x & 31;
    const int wave = threadIdx.x >> 5;
    const int t = blockIdx.x * 8 + wave;
    const float* xr = xf + (size_t)t * M_DIM;
    float s0 = 0.f, s1 = 0.f, s2 = 0.f, s3 = 0.f;
    for (int m = lane; m < M_DIM; m += 32) {
        float  xv = xr[m];
        float4 w  = *(const float4*)(wg + m * E_NUM);
        s0 += xv * w.x; s1 += xv * w.y; s2 += xv * w.z; s3 += xv * w.w;
    }
    for (int off = 16; off; off >>= 1) {
        s0 += __shfl_xor(s0, off, 32);
        s1 += __shfl_xor(s1, off, 32);
        s2 += __shfl_xor(s2, off, 32);
        s3 += __shfl_xor(s3, off, 32);
    }
    if (lane == 0) {
        float l[4] = {s0, s1, s2, s3};
        float mx = fmaxf(fmaxf(l[0], l[1]), fmaxf(l[2], l[3]));
        float p[4], sum = 0.f;
        #pragma unroll
        for (int e = 0; e < 4; ++e) { p[e] = __expf(l[e] - mx); sum += p[e]; }
        float inv = 1.f / sum;
        #pragma unroll
        for (int e = 0; e < 4; ++e) p[e] *= inv;
        int i1 = 0;
        #pragma unroll
        for (int e = 1; e < 4; ++e) if (p[e] > p[i1]) i1 = e;
        int i2 = (i1 == 0) ? 1 : 0;
        #pragma unroll
        for (int e = 0; e < 4; ++e) if (e != i1 && p[e] > p[i2]) i2 = e;
        fe[t]          = i1;  gates[t]          = p[i1];
        fe[T_TOK + t]  = i2;  gates[T_TOK + t]  = p[i2];
        #pragma unroll
        for (int e = 0; e < 4; ++e) atomicAdd(&me[e], p[e]);
    }
}

// ---------------- 2. routing: slot positions per expert ---------------------
__global__ __launch_bounds__(256) void routing_kernel(const int* __restrict__ fe,
                                                      int* __restrict__ fp,
                                                      int* __restrict__ counts) {
    __shared__ int lc[256][4];
    const int tid = threadIdx.x;
    const int per = (K_TOP * T_TOK) / 256;   // 64
    const int beg = tid * per;
    int c[4] = {0, 0, 0, 0};
    for (int i = beg; i < beg + per; ++i) c[fe[i]]++;
    #pragma unroll
    for (int e = 0; e < 4; ++e) lc[tid][e] = c[e];
    __syncthreads();
    if (tid == 0) {
        int run[4] = {0, 0, 0, 0};
        for (int t = 0; t < 256; ++t)
            #pragma unroll
            for (int e = 0; e < 4; ++e) { int v = lc[t][e]; lc[t][e] = run[e]; run[e] += v; }
        #pragma unroll
        for (int e = 0; e < 4; ++e) counts[e] = run[e];
    }
    __syncthreads();
    int base[4];
    #pragma unroll
    for (int e = 0; e < 4; ++e) base[e] = lc[tid][e];
    for (int i = beg; i < beg + per; ++i) { int e = fe[i]; fp[i] = base[e]++; }
}

// ---------------- 3. scatter tokens -> dispatched (bf16) --------------------
__global__ __launch_bounds__(256) void dispatch_kernel(const float* __restrict__ xf,
                                                       const int* __restrict__ fe,
                                                       const int* __restrict__ fp,
                                                       bf16_t* __restrict__ disp) {
    const int i = blockIdx.x;            // 0 .. 2T-1 (j-major)
    const int t = i & (T_TOK - 1);
    const int e = fe[i], p = fp[i];
    if (p >= C_CAP) return;              // dropped token
    const float* src = xf + (size_t)t * M_DIM;
    bf16_t* dst = disp + ((size_t)e * C_CAP + p) * M_DIM;
    for (int c = threadIdx.x; c < M_DIM; c += 256) dst[c] = (bf16_t)src[c];
}

// ---------------- 4. weight transpose + f32->bf16 ---------------------------
// src [R, Cc] f32 per expert -> dst [Cc, R] bf16 per expert
__global__ void transpose_cvt(const float* __restrict__ src, bf16_t* __restrict__ dst,
                              int R, int Cc, size_t sStride, size_t dStride) {
    __shared__ float tile[32][33];
    const int e = blockIdx.z;
    const float* s = src + (size_t)e * sStride;
    bf16_t* d = dst + (size_t)e * dStride;
    const int c0 = blockIdx.x * 32, r0 = blockIdx.y * 32;
    for (int i = threadIdx.y; i < 32; i += 8)
        tile[i][threadIdx.x] = s[(size_t)(r0 + i) * Cc + c0 + threadIdx.x];
    __syncthreads();
    for (int i = threadIdx.y; i < 32; i += 8)
        d[(size_t)(c0 + i) * R + r0 + threadIdx.x] = (bf16_t)tile[threadIdx.x][i];
}

// ---------------- 5. WMMA GEMM core: wave tile 32 rows x 64 cols ------------
// A [rows, K] bf16 row-major, W [cols, K] bf16 (N-major), K-loop in steps of 32.
// Per k-step: 4 A b128-loads + 8 B b128-loads -> 8 wmma (1.5 VMEM / WMMA).
// acc[r][j]: r = row sub-tile (0/1 -> +0/+16 rows), j = col sub-tile (j*16).
template <bool RELU_BF16_OUT>
__device__ inline void gemm_wave(const bf16_t* __restrict__ A, int ldA,
                                 const bf16_t* __restrict__ W, int ldW, int Kdim,
                                 const float* __restrict__ bias,
                                 void* __restrict__ outp, int ldO,
                                 int row0, int col0, int lane) {
    v8f acc[2][4];
    #pragma unroll
    for (int r = 0; r < 2; ++r)
        #pragma unroll
        for (int j = 0; j < 4; ++j)
            #pragma unroll
            for (int i = 0; i < 8; ++i) acc[r][j][i] = 0.f;

    for (int k0 = 0; k0 < Kdim; k0 += 32) {
        v16bf a0 = load_fragA(A, ldA, row0,      k0, lane);
        v16bf a1 = load_fragA(A, ldA, row0 + 16, k0, lane);
        #pragma unroll
        for (int j = 0; j < 4; ++j) {
            v16bf b = load_fragB(W, ldW, col0 + j * 16, k0, lane);
            acc[0][j] = __builtin_amdgcn_wmma_f32_16x16x32_bf16(
                false, a0, false, b, (short)0, acc[0][j], false, false);
            acc[1][j] = __builtin_amdgcn_wmma_f32_16x16x32_bf16(
                false, a1, false, b, (short)0, acc[1][j], false, false);
        }
    }

    const int hi = lane >> 4, n = lane & 15;
    #pragma unroll
    for (int j = 0; j < 4; ++j) {
        const int col = col0 + j * 16 + n;
        const float bv = bias[col];
        #pragma unroll
        for (int r = 0; r < 2; ++r) {
            #pragma unroll
            for (int v = 0; v < 8; ++v) {
                const int row = row0 + r * 16 + v + hi * 8;   // C/D layout: VGPR v -> M=v / v+8
                if (RELU_BF16_OUT) {
                    ((bf16_t*)outp)[(size_t)row * ldO + col] =
                        (bf16_t)fmaxf(acc[r][j][v] + bv, 0.f);
                } else {
                    ((float*)outp)[(size_t)row * ldO + col] = acc[r][j][v] + bv;
                }
            }
        }
    }
}

// ---------------- 5a. GEMM1: h = relu(disp @ fc1_w + b1)  bf16 out ----------
__global__ __launch_bounds__(256) void moe_gemm1(const bf16_t* __restrict__ disp,
                                                 const bf16_t* __restrict__ w1t,
                                                 const float* __restrict__ b1,
                                                 bf16_t* __restrict__ h) {
    const int e = blockIdx.z;
    const int lane = threadIdx.x & 31, wave = threadIdx.x >> 5;
    gemm_wave<true>(disp + (size_t)e * C_CAP * M_DIM, M_DIM,    // A [C, M]
                    w1t  + (size_t)e * H_DIM * M_DIM, M_DIM,    // W [H, M]
                    M_DIM,
                    b1 + (size_t)e * H_DIM,
                    h + (size_t)e * C_CAP * H_DIM, H_DIM,       // out [C, H]
                    blockIdx.y * 256 + wave * 32, blockIdx.x * 64, lane);
}

// ---------------- 5b. GEMM2: y = h @ fc2_w + b2   f32 out -------------------
__global__ __launch_bounds__(256) void moe_gemm2(const bf16_t* __restrict__ h,
                                                 const bf16_t* __restrict__ w2t,
                                                 const float* __restrict__ b2,
                                                 float* __restrict__ y) {
    const int e = blockIdx.z;
    const int lane = threadIdx.x & 31, wave = threadIdx.x >> 5;
    gemm_wave<false>(h   + (size_t)e * C_CAP * H_DIM, H_DIM,    // A [C, H]
                     w2t + (size_t)e * M_DIM * H_DIM, H_DIM,    // W [M, H]
                     H_DIM,
                     b2 + (size_t)e * M_DIM,
                     y + (size_t)e * C_CAP * M_DIM, M_DIM,      // out [C, M]
                     blockIdx.y * 256 + wave * 32, blockIdx.x * 64, lane);
}

// ---------------- 6. decode: gather + gate-weight + sum over k --------------
__global__ __launch_bounds__(256) void decode_kernel(const float* __restrict__ y,
                                                     const int* __restrict__ fe,
                                                     const int* __restrict__ fp,
                                                     const float* __restrict__ gates,
                                                     float* __restrict__ out) {
    const int t = blockIdx.x;
    float acc[4] = {0.f, 0.f, 0.f, 0.f};   // 1024 cols / 256 threads
    #pragma unroll
    for (int j = 0; j < K_TOP; ++j) {
        const int i = j * T_TOK + t;
        const int e = fe[i], p = fp[i];
        if (p < C_CAP) {
            const float g = gates[i];
            const float* yr = y + ((size_t)e * C_CAP + p) * M_DIM;
            #pragma unroll
            for (int c = 0; c < 4; ++c) acc[c] += g * yr[threadIdx.x + c * 256];
        }
    }
    #pragma unroll
    for (int c = 0; c < 4; ++c)
        out[(size_t)t * M_DIM + threadIdx.x + c * 256] = acc[c];
}

// ---------------- 7. l_aux ---------------------------------------------------
__global__ void laux_kernel(const float* __restrict__ me, const int* __restrict__ counts,
                            float* __restrict__ out_laux) {
    float s = 0.f;
    #pragma unroll
    for (int e = 0; e < E_NUM; ++e)
        s += (me[e] / (float)T_TOK) * ((float)counts[e] / (float)(T_TOK * K_TOP));
    *out_laux = (float)E_NUM * s;
}

// ---------------------------------------------------------------------------
extern "C" void kernel_launch(void* const* d_in, const int* in_sizes, int n_in,
                              void* d_out, int out_size, void* d_ws, size_t ws_size,
                              hipStream_t stream) {
    const float* x     = (const float*)d_in[0];   // [4,2048,1024]
    const float* wg    = (const float*)d_in[1];   // [1024,4]
    const float* fc1_w = (const float*)d_in[2];   // [4,1024,4096]
    const float* fc1_b = (const float*)d_in[3];   // [4,1,4096]
    const float* fc2_w = (const float*)d_in[4];   // [4,4096,1024]
    const float* fc2_b = (const float*)d_in[5];   // [4,1,1024]
    float* out = (float*)d_out;                   // [T*M] tokens + 1 (l_aux)

    // ---- workspace layout ----
    char* ws = (char*)d_ws;
    size_t off = 0;
    auto alloc = [&](size_t bytes) { void* p = ws + off; off = (off + bytes + 255) & ~(size_t)255; return p; };
    bf16_t* w1t   = (bf16_t*)alloc((size_t)E_NUM * H_DIM * M_DIM * 2);  // [E,H,M]
    bf16_t* w2t   = (bf16_t*)alloc((size_t)E_NUM * M_DIM * H_DIM * 2);  // [E,M,H]
    bf16_t* disp  = (bf16_t*)alloc((size_t)E_NUM * C_CAP * M_DIM * 2);  // [E,C,M]
    bf16_t* hbuf  = (bf16_t*)alloc((size_t)E_NUM * C_CAP * H_DIM * 2);  // [E,C,H]
    float*  ybuf  = (float*) alloc((size_t)E_NUM * C_CAP * M_DIM * 4);  // [E,C,M]
    int*    fe    = (int*)   alloc((size_t)K_TOP * T_TOK * 4);
    int*    fp    = (int*)   alloc((size_t)K_TOP * T_TOK * 4);
    float*  gates = (float*) alloc((size_t)K_TOP * T_TOK * 4);
    float*  me    = (float*) alloc(E_NUM * 4);
    int*    cnts  = (int*)   alloc(E_NUM * 4);
    (void)ws_size; (void)in_sizes; (void)n_in; (void)out_size;

    // ---- zero accumulators & dispatch buffer (graph-capture safe) ----
    hipMemsetAsync(me,   0, E_NUM * 4, stream);
    hipMemsetAsync(cnts, 0, E_NUM * 4, stream);
    hipMemsetAsync(disp, 0, (size_t)E_NUM * C_CAP * M_DIM * 2, stream);

    // ---- weight transpose + convert ----
    transpose_cvt<<<dim3(H_DIM / 32, M_DIM / 32, E_NUM), dim3(32, 8), 0, stream>>>(
        fc1_w, w1t, M_DIM, H_DIM, (size_t)M_DIM * H_DIM, (size_t)H_DIM * M_DIM);
    transpose_cvt<<<dim3(M_DIM / 32, H_DIM / 32, E_NUM), dim3(32, 8), 0, stream>>>(
        fc2_w, w2t, H_DIM, M_DIM, (size_t)H_DIM * M_DIM, (size_t)M_DIM * H_DIM);

    // ---- gating / routing / dispatch ----
    gate_kernel<<<T_TOK / 8, 256, 0, stream>>>(x, wg, fe, gates, me);
    routing_kernel<<<1, 256, 0, stream>>>(fe, fp, cnts);
    dispatch_kernel<<<K_TOP * T_TOK, 256, 0, stream>>>(x, fe, fp, disp);

    // ---- expert FFN (WMMA bf16), wave tile 32x64, block tile 256x64 --------
    moe_gemm1<<<dim3(H_DIM / 64, C_CAP / 256, E_NUM), 256, 0, stream>>>(disp, w1t, fc1_b, hbuf);
    moe_gemm2<<<dim3(M_DIM / 64, C_CAP / 256, E_NUM), 256, 0, stream>>>(hbuf, w2t, fc2_b, ybuf);

    // ---- decode + aux loss ----
    decode_kernel<<<T_TOK, 256, 0, stream>>>(ybuf, fe, fp, gates, out);
    laux_kernel<<<1, 1, 0, stream>>>(me, cnts, out + (size_t)T_TOK * M_DIM);
}
